// QuadNeighborhoodEncoderAttention_13142599926343
// MI455X (gfx1250) — matrix operand order repaired and checked by reading, not compiled, not run on hardware
//
#include <hip/hip_runtime.h>
#include <hip/hip_bf16.h>

// ---------------------------------------------------------------------------
// QuadNeighborhoodEncoderAttention for MI455X (gfx1250, wave32, WMMA)
//
// Persistent blocks: all f16 weights (252 KB padded, incl. a_W3 padded to
// 128x16) staged into LDS once per block (TDM tensor_load_to_lds when
// available), then each of the 4 waves loops over 16-row tiles
// (2 batch x 8 neighbors). All GEMMs -- including the attention score
// projection -- use V_WMMA_F32_16X16X32_F16 with LDS-resident B fragments.
// Activations ping-pong through bank-conflict-padded wave-private LDS.
// Activation = hardware V_TANH_F32 (TRANS op, co-executes with WMMA).
// ---------------------------------------------------------------------------

typedef __attribute__((ext_vector_type(16))) _Float16 v16h;
typedef __attribute__((ext_vector_type(8)))  _Float16 v8h;
typedef __attribute__((ext_vector_type(8)))  float    v8f;
typedef __attribute__((ext_vector_type(4)))  unsigned int v4u;
typedef __attribute__((ext_vector_type(8)))  int      v8i;
typedef __attribute__((ext_vector_type(4)))  int      v4i;

#define HID 128
#define SELF_DIM 18
#define NBR_DIM 6
#define OBS_W 66            // 18 + 8*6
#define NNBR 8

// padded strides (halves) -> lane rows land 4 banks apart (conflict-free
// per 16-lane half-wave for b128 LDS accesses)
#define WSTR  136           // weight row stride (128 cols + 8 pad)
#define B0STR 264           // buf0 stride (256 cols + 8 pad)
#define INSTR 40            // first-layer input stride (32 cols + 8 pad)

// weight rows: e1 32 | e2 128 | v1 128 | v2 128 | a1 256 | a2 128 | a3 128
#define OFF_E1 0
#define OFF_E2 (32  * WSTR)
#define OFF_V1 (160 * WSTR)
#define OFF_V2 (288 * WSTR)
#define OFF_A1 (416 * WSTR)
#define OFF_A2 (672 * WSTR)
#define OFF_A3 (800 * WSTR)
#define WGT_HALVES (928 * WSTR)            // 126208 halves = 252416 B

#define ACT_PER_WAVE (16 * B0STR + 2 * (16 * WSTR))  // 4224 + 4352 = 8576
#define LDS_HALVES (WGT_HALVES + 4 * ACT_PER_WAVE)   // 160512 halves = 321024 B

#if defined(__has_include)
#  if __has_include(<hip/amd_detail/amd_gfx1250_TDM.h>)
#    define TDM_6ARG 1
#  endif
#endif
#if __has_builtin(__builtin_amdgcn_tensor_load_to_lds)
#  define HAVE_TDM 1
#endif

static __device__ __forceinline__ void lds_fence() {
    asm volatile("s_wait_dscnt 0" ::: "memory");
}

// Hardware tanh (CDNA5 TRANS op).
static __device__ __forceinline__ float htanh(float x) {
#if __has_builtin(__builtin_amdgcn_tanhf)
    return __builtin_amdgcn_tanhf(x);
#else
    float y;
    asm volatile("v_tanh_f32 %0, %1" : "=v"(y) : "v"(x));
    return y;
#endif
}

static __device__ __forceinline__ v16h pack16(v8h lo, v8h hi) {
    v16h r;
#pragma unroll
    for (int i = 0; i < 8; ++i) { r[i] = lo[i]; r[i + 8] = hi[i]; }
    return r;
}

// Load an A fragment (16x32 f16) for K-chunk kc from padded LDS tile.
static __device__ __forceinline__ v16h load_afrag(
    const _Float16* __restrict__ inL, int inStride, int kc, int lane)
{
    const int M = lane & 15, sel = lane >> 4;
    const _Float16* p = inL + M * inStride + kc * 32 + sel * 8;
    return pack16(*(const v8h*)p, *(const v8h*)(p + 16));
}

// One dense layer for a 16-row tile: out[16x128] = tanh(in[16x(32*KC)] @ W + b)
template <int KC>
static __device__ __forceinline__ void mlp_layer(
    const _Float16* __restrict__ inL, int inStride,
    const _Float16* __restrict__ Wlds,
    const float* __restrict__ bias,
    _Float16* __restrict__ outL, int outStride,
    int lane)
{
    const int M   = lane & 15;
    const int sel = lane >> 4;

    v16h A[KC];
#pragma unroll
    for (int kc = 0; kc < KC; ++kc) A[kc] = load_afrag(inL, inStride, kc, lane);

#pragma unroll
    for (int j = 0; j < 8; ++j) {           // 8 N-blocks of 16 columns
        v8f acc = {};
#pragma unroll
        for (int kc = 0; kc < KC; ++kc) {
            const _Float16* wp = Wlds + (kc * 32 + sel * 8) * WSTR + j * 16 + M;
            v16h Bf;
#pragma unroll
            for (int s = 0; s < 8; ++s) { Bf[s] = wp[s * WSTR]; Bf[s + 8] = wp[(s + 16) * WSTR]; }
            acc = __builtin_amdgcn_wmma_f32_16x16x32_f16(
                false, A[kc], false, Bf, (short)0, acc, false, false);
        }
        const int   N  = j * 16 + M;
        const float bv = bias[N];
#pragma unroll
        for (int r = 0; r < 8; ++r) {
            float v = htanh(acc[r] + bv);
            outL[(r + sel * 8) * outStride + N] = (_Float16)v;
        }
    }
}

__global__ __launch_bounds__(128) void quad_attn_kernel(
    const float* __restrict__ self_obs,   // [B,18]
    const float* __restrict__ obs,        // [B,66]
    const _Float16* __restrict__ wg,      // padded f16 weights in ws
    const float* __restrict__ eb1, const float* __restrict__ eb2,
    const float* __restrict__ vb1, const float* __restrict__ vb2,
    const float* __restrict__ ab1, const float* __restrict__ ab2,
    const float* __restrict__ ab3,
    float* __restrict__ out,              // [B,128]
    int B, int R)
{
    extern __shared__ _Float16 smem[];
    const int lane = threadIdx.x & 31;
    const int wave = threadIdx.x >> 5;

    // ---- stage all weights into LDS once per block -------------------------
#if defined(HAVE_TDM)
    if (wave == 0) {
        unsigned long long ga = (unsigned long long)(const void*)wg;
        // D# group0: count=1 | lds_addr=0 | global_addr | type=2
        v4u g0 = { 1u, 0u, (unsigned)(ga & 0xFFFFFFFFu),
                   (unsigned)(((ga >> 32) & 0x01FFFFFFu) | 0x80000000u) };
        // D# group1: data_size=8B, tensor_dim0=tile_dim0=stride=31552 (x8B),
        // tensor_dim1=tile_dim1=1  (flat 252416-byte 1-D copy)
        v8i g1 = { 0x30000, 0x7B400000, 0x00010000, 0x7B400000,
                   1, 31552, 0, 0 };
        v4i z4 = { 0, 0, 0, 0 };
#if defined(TDM_6ARG)
        v8i z8 = { 0, 0, 0, 0, 0, 0, 0, 0 };
        __builtin_amdgcn_tensor_load_to_lds(g0, g1, z4, z4, z8, 0);
#else
        __builtin_amdgcn_tensor_load_to_lds(g0, g1, z4, z4, 0);
#endif
#if __has_builtin(__builtin_amdgcn_s_wait_tensorcnt)
        __builtin_amdgcn_s_wait_tensorcnt((short)0);
#else
        asm volatile("s_wait_tensorcnt 0" ::: "memory");
#endif
    }
#else
    {
        const v8h* g = (const v8h*)wg;
        v8h*       s = (v8h*)smem;
        for (int i = threadIdx.x; i < WGT_HALVES / 8; i += 128) s[i] = g[i];
    }
#endif
    __syncthreads();

    const _Float16* wlds = smem;
    _Float16* act  = smem + WGT_HALVES + wave * ACT_PER_WAVE;
    _Float16* buf0 = act;                 // 16 x 264 (emb | mean -> attn input)
    _Float16* buf1 = buf0 + 16 * B0STR;   // 16 x 136 scratch (+ layer-1 input)
    _Float16* vals = buf1 + 16 * WSTR;    // 16 x 136 value-MLP output

    const int numTiles = R >> 4;          // 16-row tiles
    for (int t = blockIdx.x * 4 + wave; t < numTiles; t += gridDim.x * 4) {
        const int rowBase = t << 4;
        const int baseSelf = rowBase % B; // one scalar modulo per tile

        // ---- stage layer-1 input tile [16 x 32] (24 valid K, zero pad) -----
        for (int idx = lane; idx < 16 * 32; idx += 32) {
            int m = idx >> 5, c = idx & 31;
            float v = 0.f;
            if (c < SELF_DIM) {
                int sr = baseSelf + m; if (sr >= B) sr -= B;  // (rowBase+m)%B
                v = self_obs[(size_t)sr * SELF_DIM + c];
            } else if (c < SELF_DIM + NBR_DIM) {
                int r = rowBase + m;
                int b = r >> 3, k = r & 7;
                v = obs[(size_t)b * OBS_W + SELF_DIM + k * NBR_DIM + (c - SELF_DIM)];
            }
            buf1[m * INSTR + c] = (_Float16)v;
        }
        lds_fence();

        // ---- embedding MLP: 24->128->128 -----------------------------------
        mlp_layer<1>(buf1, INSTR, wlds + OFF_E1, eb1, vals, WSTR, lane);
        lds_fence();
        mlp_layer<4>(vals, WSTR, wlds + OFF_E2, eb2, buf0, B0STR, lane); // emb
        lds_fence();

        // ---- neighbor mean (packed f16) -> buf0[:,128:256] -----------------
        {
            int g  = lane >> 4;           // batch-half of the tile
            int c0 = (lane & 15) * 8;     // 8-column strip per lane
            v8h sum = {};
#pragma unroll
            for (int m8 = 0; m8 < 8; ++m8)
                sum += *(const v8h*)(buf0 + (g * 8 + m8) * B0STR + c0);
            v8h mean = sum * (_Float16)0.125f;
#pragma unroll
            for (int m8 = 0; m8 < 8; ++m8)
                *(v8h*)(buf0 + (g * 8 + m8) * B0STR + 128 + c0) = mean;
        }
        lds_fence();

        // ---- value MLP: emb -> 128 -> 128 ----------------------------------
        mlp_layer<4>(buf0, B0STR, wlds + OFF_V1, vb1, buf1, WSTR, lane);
        lds_fence();
        mlp_layer<4>(buf1, WSTR,  wlds + OFF_V2, vb2, vals, WSTR, lane);
        lds_fence();

        // ---- attention MLP: [emb|mean](256) -> 128 -> 128 ------------------
        mlp_layer<8>(buf0, B0STR, wlds + OFF_A1, ab1, buf1, WSTR, lane);
        lds_fence();
        mlp_layer<4>(buf1, WSTR,  wlds + OFF_A2, ab2, buf1, WSTR, lane);
        lds_fence();

        // ---- scores via WMMA: h2[16x128] @ a3pad[128x16], col 0 = scores ---
        {
            v16h A[4];
#pragma unroll
            for (int kc = 0; kc < 4; ++kc) A[kc] = load_afrag(buf1, WSTR, kc, lane);
            v8f acc = {};
#pragma unroll
            for (int kc = 0; kc < 4; ++kc) {
                const _Float16* wp = wlds + OFF_A3 + (kc * 32 + (lane >> 4) * 8) * WSTR + (lane & 15);
                v16h Bf;
#pragma unroll
                for (int s = 0; s < 8; ++s) { Bf[s] = wp[s * WSTR]; Bf[s + 8] = wp[(s + 16) * WSTR]; }
                acc = __builtin_amdgcn_wmma_f32_16x16x32_f16(
                    false, A[kc], false, Bf, (short)0, acc, false, false);
            }
            // C layout: N=0 lives in lanes 0 (M=0..7) and 16 (M=8..15)
            float* scr = (float*)buf0;    // buf0 free after a1
            if ((lane & 15) == 0) {
                const int sel = lane >> 4;
#pragma unroll
                for (int r = 0; r < 8; ++r) scr[r + sel * 8] = acc[r];
            }
        }
        lds_fence();

        float sc = ((const float*)buf0)[lane & 15] + ab3[0];
        float mx = sc;
        mx = fmaxf(mx, __shfl_xor(mx, 1));
        mx = fmaxf(mx, __shfl_xor(mx, 2));
        mx = fmaxf(mx, __shfl_xor(mx, 4));
        float e = __expf(sc - mx);
        float sum = e;
        sum += __shfl_xor(sum, 1);
        sum += __shfl_xor(sum, 2);
        sum += __shfl_xor(sum, 4);
        float attn = e / sum;             // lane m<16 holds attn of row m

        // ---- out[b] = sum_k attn[k] * vals[k] (2 batch rows per wave) ------
#pragma unroll
        for (int bi = 0; bi < 2; ++bi) {
            const int c0 = lane * 4;
            float a0 = 0.f, a1 = 0.f, a2 = 0.f, a3 = 0.f;
#pragma unroll
            for (int k = 0; k < 8; ++k) {
                float ak = __shfl(attn, bi * 8 + k);
                const _Float16* vp = vals + (bi * 8 + k) * WSTR + c0;
                a0 += ak * (float)vp[0];
                a1 += ak * (float)vp[1];
                a2 += ak * (float)vp[2];
                a3 += ak * (float)vp[3];
            }
            size_t b = (size_t)(rowBase >> 3) + bi;
            float4 o = make_float4(a0, a1, a2, a3);
            *(float4*)(out + b * HID + c0) = o;
        }
        lds_fence();   // vals/scr consumed before next iteration overwrites
    }
}

// ---------------------------------------------------------------------------
// Weight pre-pass: fp32 -> f16, padded rows (WSTR) in workspace.
// ---------------------------------------------------------------------------
__global__ void cvt_pad_rows(const float* __restrict__ src,
                             _Float16* __restrict__ dst,
                             int Kvalid, int Ktot) {
    for (int i = blockIdx.x * blockDim.x + threadIdx.x; i < Ktot * HID;
         i += gridDim.x * blockDim.x) {
        int k = i >> 7, c = i & 127;
        dst[k * WSTR + c] =
            (k < Kvalid) ? (_Float16)src[k * HID + c] : (_Float16)0.f;
    }
}

// a_W3 [128x1] -> padded [128x16] block: col 0 = weights, cols 1..15 = 0
__global__ void cvt_a3(const float* __restrict__ src,
                       _Float16* __restrict__ dst) {
    int i = blockIdx.x * blockDim.x + threadIdx.x;
    if (i < 128 * 16) {
        int k = i >> 4, c = i & 15;
        dst[k * WSTR + c] = (c == 0) ? (_Float16)src[k] : (_Float16)0.f;
    }
}

extern "C" void kernel_launch(void* const* d_in, const int* in_sizes, int n_in,
                              void* d_out, int out_size, void* d_ws, size_t ws_size,
                              hipStream_t stream) {
    const float* self_obs = (const float*)d_in[0];
    const float* obs      = (const float*)d_in[1];
    const float* eW1 = (const float*)d_in[2];  const float* eb1 = (const float*)d_in[3];
    const float* eW2 = (const float*)d_in[4];  const float* eb2 = (const float*)d_in[5];
    const float* vW1 = (const float*)d_in[6];  const float* vb1 = (const float*)d_in[7];
    const float* vW2 = (const float*)d_in[8];  const float* vb2 = (const float*)d_in[9];
    const float* aW1 = (const float*)d_in[10]; const float* ab1 = (const float*)d_in[11];
    const float* aW2 = (const float*)d_in[12]; const float* ab2 = (const float*)d_in[13];
    const float* aW3 = (const float*)d_in[14]; const float* ab3 = (const float*)d_in[15];

    const int B = in_sizes[0] / SELF_DIM;
    const int R = B * NNBR;

    _Float16* ws = (_Float16*)d_ws;
    cvt_pad_rows<<<16,  256, 0, stream>>>(eW1, ws + OFF_E1, 24, 32);
    cvt_pad_rows<<<64,  256, 0, stream>>>(eW2, ws + OFF_E2, 128, 128);
    cvt_pad_rows<<<64,  256, 0, stream>>>(vW1, ws + OFF_V1, 128, 128);
    cvt_pad_rows<<<64,  256, 0, stream>>>(vW2, ws + OFF_V2, 128, 128);
    cvt_pad_rows<<<128, 256, 0, stream>>>(aW1, ws + OFF_A1, 256, 256);
    cvt_pad_rows<<<64,  256, 0, stream>>>(aW2, ws + OFF_A2, 128, 128);
    cvt_a3      <<<8,   256, 0, stream>>>(aW3, ws + OFF_A3);

    const int numTiles = R >> 4;                 // 32768 for B=65536
    int blocks = 512;                            // persistent blocks (4 waves)
    if (blocks * 4 > numTiles) blocks = (numTiles + 3) / 4;
    const size_t shmem = (size_t)LDS_HALVES * sizeof(_Float16);  // 321024 B
    quad_attn_kernel<<<blocks, 128, shmem, stream>>>(
        self_obs, obs, ws, eb1, eb2, vb1, vb2, ab1, ab2, ab3,
        (float*)d_out, B, R);
}
